// HFFM_10788957847660
// MI455X (gfx1250) — matrix operand order repaired and checked by reading, not compiled, not run on hardware
//
#include <hip/hip_runtime.h>

// Problem constants (reference: B=16, C=512, H=W=64)
#define BB 16
#define CC 512
#define HWN 4096
#define LN_EPS 1e-5f

typedef __attribute__((ext_vector_type(16))) __bf16 v16bf;
typedef __attribute__((ext_vector_type(8)))  float  v8f;

union BF16Frag { v16bf v; uint4 q[2]; };

// round-to-nearest-even f32 -> bf16 (as raw u16)
static __device__ __forceinline__ unsigned short f2bf(float f) {
    unsigned int u = __float_as_uint(f);
    u += 0x7FFFu + ((u >> 16) & 1u);
    return (unsigned short)(u >> 16);
}

// ---------------------------------------------------------------------------
// K1: convert x (f32 [B,C,HW]) -> vbf (bf16 [B,C,HW]) and vt (bf16 [B,HW,C])
// 32x32 LDS tile transpose; block (32,8), grid (HW/32, C/32, B)
// ---------------------------------------------------------------------------
__global__ void k_cvt(const float* __restrict__ x,
                      unsigned short* __restrict__ vbf,
                      unsigned short* __restrict__ vt) {
    __shared__ unsigned short tile[32][33];
    const int b  = blockIdx.z;
    const int n0 = blockIdx.x * 32;
    const int c0 = blockIdx.y * 32;
    const size_t bofs = (size_t)b * CC * HWN;
    const float* xb = x + bofs;
    unsigned short* vb = vbf + bofs;

    for (int i = threadIdx.y; i < 32; i += 8) {
        const int c = c0 + i;
        const int n = n0 + threadIdx.x;
        const unsigned short h = f2bf(xb[(size_t)c * HWN + n]);
        vb[(size_t)c * HWN + n] = h;
        tile[i][threadIdx.x] = h;
    }
    __syncthreads();
    unsigned short* vtb = vt + (size_t)b * HWN * CC;
    for (int i = threadIdx.y; i < 32; i += 8) {
        const int n = n0 + i;
        const int c = c0 + threadIdx.x;
        vtb[(size_t)n * CC + c] = tile[threadIdx.x][i];
    }
}

// ---------------------------------------------------------------------------
// K2: energy[b] = v[b] * v[b]^T   (M=N=C=512, K=HW=4096), bf16 WMMA, f32 acc
// One wave per 64x32 output block (4x2 tiles). Single base pointer per
// operand; tile displacements are constant immediate offsets.
// block 32, grid (C/32, C/64, B)
// ---------------------------------------------------------------------------
__global__ void k_gemm1(const unsigned short* __restrict__ vbf,
                        float* __restrict__ energy) {
    const int b  = blockIdx.z;
    const int i0 = blockIdx.y * 64;
    const int j0 = blockIdx.x * 32;
    const int lane   = threadIdx.x & 31;
    const int laneLo = lane & 15;
    const int laneHi = lane >> 4;

    const unsigned short* base = vbf + (size_t)b * CC * HWN;
    // A frag base: row i0+laneLo, K chunks {0..7,16..23} (+8 for hi lanes)
    const unsigned short* aB = base + (size_t)(i0 + laneLo) * HWN + (laneHi ? 8 : 0);
    // B frag base: column j of B == row j of v, K chunks {0..7,16..23} (+8 for hi lanes)
    const unsigned short* bB = base + (size_t)(j0 + laneLo) * HWN + (laneHi ? 8 : 0);

    v8f acc[4][2] = {};
    for (int k = 0; k < HWN; k += 32) {
        BF16Frag A[4], Bf[2];
        #pragma unroll
        for (int m = 0; m < 4; ++m) {
            A[m].q[0] = *(const uint4*)(aB + k + m * 16 * HWN);
            A[m].q[1] = *(const uint4*)(aB + k + m * 16 * HWN + 16);
        }
        #pragma unroll
        for (int n = 0; n < 2; ++n) {
            Bf[n].q[0] = *(const uint4*)(bB + k + n * 16 * HWN);
            Bf[n].q[1] = *(const uint4*)(bB + k + n * 16 * HWN + 16);
        }
        #pragma unroll
        for (int m = 0; m < 4; ++m)
            #pragma unroll
            for (int n = 0; n < 2; ++n)
                acc[m][n] = __builtin_amdgcn_wmma_f32_16x16x32_bf16(
                    false, A[m].v, false, Bf[n].v, (short)0, acc[m][n], false, false);
    }

    float* e = energy + (size_t)b * CC * CC;
    const int N = laneLo;
    #pragma unroll
    for (int m = 0; m < 4; ++m)
        #pragma unroll
        for (int n = 0; n < 2; ++n)
            #pragma unroll
            for (int r = 0; r < 8; ++r) {
                const int M = r + 8 * laneHi;
                e[(size_t)(i0 + 16 * m + M) * CC + (j0 + 16 * n + N)] = acc[m][n][r];
            }
}

// ---------------------------------------------------------------------------
// K3: row softmax over energy [B*C, C], write bf16 attn. warp per row.
// block 128 (4 warps), grid B*C/4
// ---------------------------------------------------------------------------
__global__ void k_softmax(const float* __restrict__ energy,
                          unsigned short* __restrict__ attn) {
    const int warp = threadIdx.x >> 5;
    const int lane = threadIdx.x & 31;
    const int row  = blockIdx.x * 4 + warp;   // over B*C
    const float* e = energy + (size_t)row * CC;

    float vals[16];
    float m = -3.0e38f;
    #pragma unroll
    for (int i = 0; i < 16; ++i) { vals[i] = e[lane + 32 * i]; m = fmaxf(m, vals[i]); }
    #pragma unroll
    for (int o = 16; o > 0; o >>= 1) m = fmaxf(m, __shfl_xor(m, o, 32));
    float s = 0.f;
    #pragma unroll
    for (int i = 0; i < 16; ++i) { vals[i] = __expf(vals[i] - m); s += vals[i]; }
    #pragma unroll
    for (int o = 16; o > 0; o >>= 1) s += __shfl_xor(s, o, 32);
    const float inv = 1.0f / s;
    unsigned short* a = attn + (size_t)row * CC;
    #pragma unroll
    for (int i = 0; i < 16; ++i) a[lane + 32 * i] = f2bf(vals[i] * inv);
}

// ---------------------------------------------------------------------------
// K4: out[b] = gamma * (attn[b] * v[b]) + x[b]   (M=C, N=HW, K=C)
// A = attn (bf16 row-major), B K-contiguous via vt (bf16 [B,HW,C]).
// One wave per 64x64 block (4x4 tiles). Single base pointer per operand;
// tile displacements are constant immediate offsets (16*CC*2 = 16 KB apart).
// block 32, grid (HW/64, C/64, B)
// ---------------------------------------------------------------------------
__global__ void k_gemm2(const unsigned short* __restrict__ attn,
                        const unsigned short* __restrict__ vt,
                        const float* __restrict__ x,
                        const float* __restrict__ gamma,
                        float* __restrict__ out) {
    const int b  = blockIdx.z;
    const int m0 = blockIdx.y * 64;
    const int n0 = blockIdx.x * 64;
    const int lane   = threadIdx.x & 31;
    const int laneLo = lane & 15;
    const int laneHi = lane >> 4;

    const unsigned short* aB = attn + (size_t)b * CC * CC
                             + (size_t)(m0 + laneLo) * CC + (laneHi ? 8 : 0);
    const unsigned short* bB = vt + (size_t)b * HWN * CC
                             + (size_t)(n0 + laneLo) * CC + (laneHi ? 8 : 0);

    v8f acc[4][4] = {};
    for (int k = 0; k < CC; k += 32) {
        BF16Frag A[4], Bf[4];
        #pragma unroll
        for (int m = 0; m < 4; ++m) {
            A[m].q[0] = *(const uint4*)(aB + k + m * 16 * CC);
            A[m].q[1] = *(const uint4*)(aB + k + m * 16 * CC + 16);
        }
        #pragma unroll
        for (int n = 0; n < 4; ++n) {
            Bf[n].q[0] = *(const uint4*)(bB + k + n * 16 * CC);
            Bf[n].q[1] = *(const uint4*)(bB + k + n * 16 * CC + 16);
        }
        #pragma unroll
        for (int m = 0; m < 4; ++m)
            #pragma unroll
            for (int n = 0; n < 4; ++n)
                acc[m][n] = __builtin_amdgcn_wmma_f32_16x16x32_bf16(
                    false, A[m].v, false, Bf[n].v, (short)0, acc[m][n], false, false);
    }

    const float g = gamma[0];
    const float* xb = x + (size_t)b * CC * HWN;
    float* ob = out + (size_t)b * CC * HWN;
    const int N = laneLo;
    #pragma unroll
    for (int m = 0; m < 4; ++m)
        #pragma unroll
        for (int r = 0; r < 8; ++r) {
            const int M = r + 8 * laneHi;
            const size_t rowo = (size_t)(m0 + 16 * m + M) * HWN + n0;
            #pragma unroll
            for (int n = 0; n < 4; ++n) {
                const size_t i = rowo + 16 * n + N;
                ob[i] = g * acc[m][n][r] + xb[i];
            }
        }
}

// ---------------------------------------------------------------------------
// K5: in-place LayerNorm over last dim (HW=4096) per row. block 256, grid B*C
// ---------------------------------------------------------------------------
__global__ void k_ln(float* __restrict__ out,
                     const float* __restrict__ w,
                     const float* __restrict__ bi) {
    const int row = blockIdx.x;             // over B*C
    float* p = out + (size_t)row * HWN;

    float s = 0.f, s2 = 0.f;
    for (int i = threadIdx.x; i < HWN; i += 256) {
        const float v = p[i];
        s += v; s2 += v * v;
    }
    #pragma unroll
    for (int o = 16; o > 0; o >>= 1) { s += __shfl_xor(s, o, 32); s2 += __shfl_xor(s2, o, 32); }

    __shared__ float shs[8], shs2[8];
    __shared__ float fmean, finv;
    const int warp = threadIdx.x >> 5;
    const int lane = threadIdx.x & 31;
    if (lane == 0) { shs[warp] = s; shs2[warp] = s2; }
    __syncthreads();
    if (threadIdx.x == 0) {
        float S = 0.f, S2 = 0.f;
        #pragma unroll
        for (int i = 0; i < 8; ++i) { S += shs[i]; S2 += shs2[i]; }
        const float mean = S * (1.0f / HWN);
        const float var  = S2 * (1.0f / HWN) - mean * mean;
        fmean = mean;
        finv  = rsqrtf(var + LN_EPS);
    }
    __syncthreads();
    const float mean = fmean, inv = finv;
    for (int i = threadIdx.x; i < HWN; i += 256) {
        p[i] = (p[i] - mean) * inv * w[i] + bi[i];
    }
}

// ---------------------------------------------------------------------------
extern "C" void kernel_launch(void* const* d_in, const int* in_sizes, int n_in,
                              void* d_out, int out_size, void* d_ws, size_t ws_size,
                              hipStream_t stream) {
    const float* x      = (const float*)d_in[0];
    const float* gamma  = (const float*)d_in[1];
    const float* ln_w   = (const float*)d_in[2];
    const float* ln_b   = (const float*)d_in[3];
    float*       out    = (float*)d_out;

    // Workspace layout (~152 MB total)
    unsigned short* vbf = (unsigned short*)d_ws;                       // [B,C,HW] bf16
    unsigned short* vt  = vbf + (size_t)BB * CC * HWN;                 // [B,HW,C] bf16
    float*          energy = (float*)(vt + (size_t)BB * CC * HWN);     // [B,C,C]  f32
    unsigned short* attn   = (unsigned short*)(energy + (size_t)BB * CC * CC); // [B,C,C] bf16

    // 1) f32 -> bf16 convert + transpose
    k_cvt<<<dim3(HWN / 32, CC / 32, BB), dim3(32, 8), 0, stream>>>(x, vbf, vt);
    // 2) energy = v v^T (bf16 WMMA, f32 acc), 64x32 per wave
    k_gemm1<<<dim3(CC / 32, CC / 64, BB), dim3(32), 0, stream>>>(vbf, energy);
    // 3) row softmax -> bf16 attn
    k_softmax<<<dim3((BB * CC) / 4), dim3(128), 0, stream>>>(energy, attn);
    // 4) out = gamma * (attn v) + x (bf16 WMMA, f32 acc, fused residual), 64x64 per wave
    k_gemm2<<<dim3(HWN / 64, CC / 64, BB), dim3(32), 0, stream>>>(attn, vt, x, gamma, out);
    // 5) in-place LayerNorm over HW
    k_ln<<<dim3(BB * CC), dim3(256), 0, stream>>>(out, ln_w, ln_b);
}
